// XLinearAttention_37563783970977
// MI455X (gfx1250) — matrix operand order repaired and checked by reading, not compiled, not run on hardware
//
#include <hip/hip_runtime.h>

// Problem constants
#define NB   4
#define NQ   128
#define NV   128
#define ND   512

typedef __attribute__((ext_vector_type(16))) __bf16 v16bf;
typedef __attribute__((ext_vector_type(8)))  float  v8f;
typedef __attribute__((ext_vector_type(4)))  int    v4i;

union AV { v16bf v; __bf16 e[16]; v4i i[2]; };
union F4 { float4 v; float f[4]; };

__device__ __forceinline__ v8f v8f_zero() {
  v8f z;
#pragma unroll
  for (int i = 0; i < 8; ++i) z[i] = 0.f;
  return z;
}

__device__ __forceinline__ v8f wmma_bf16(const AV& a, const AV& b, v8f c) {
  return __builtin_amdgcn_wmma_f32_16x16x32_bf16(false, a.v, false, b.v,
                                                 (short)0, c, false, false);
}

// ---------------------------------------------------------------------------
// Kernel A: convert W1 / Wse (f32 row-major [k=512][n=512]) into bf16 tiles in
// WMMA B-operand layout: tile = (kt*32 + ntile), within tile: lane l holds
// column n = ntile*16 + (l&15); elem j holds k = kt*32 + (l>>4)*16 + j.
// Each tile = 32 lanes * 16 bf16 = 1 KB, contiguous per lane (two b128 loads).
// ---------------------------------------------------------------------------
__global__ __launch_bounds__(256) void prep_weights(
    const float* __restrict__ W1, const float* __restrict__ Wse,
    __bf16* __restrict__ w1t, __bf16* __restrict__ wset)
{
  int gid  = blockIdx.x * 256 + threadIdx.x;   // 0 .. 2*512*512-1
  int m    = gid >> 18;                        // 0 = W1, 1 = Wse
  int r    = gid & 262143;
  int tile = r >> 9;                           // 0..511 (kt*32 + ntile)
  int tv   = r & 511;                          // lane*16 + j
  int lane = tv >> 4;
  int j    = tv & 15;
  int kt   = tile >> 5;
  int ntl  = tile & 31;
  int kk   = kt * 32 + ((lane >> 4) << 4) + j;
  int col  = ntl * 16 + (lane & 15);
  const float* src = m ? Wse : W1;
  __bf16*      dst = m ? wset : w1t;
  dst[(tile << 9) + tv] = (__bf16)src[kk * ND + col];
}

// ---------------------------------------------------------------------------
// Kernel B: one workgroup per (b,q). 256 threads = 8 waves; wave w owns
// v-rows [16w, 16w+16). A-operand (X = q1 .* ks, bf16) built once into
// registers (16 K-tiles * 8 VGPR). N looped in 4 blocks of 128 cols.
// The W1 bf16 panel is DOUBLE-BUFFERED in LDS (2 x 128 KB): while computing
// on panel[cur], the next panel's global->LDS traffic (2 x b128 per thread
// per kt step) is interleaved into the WMMA loop, so the L2 fill overlaps the
// matrix stream instead of being barrier-serialized.
// Epilogue fuses relu(+b1), h@W2 partials (-> s), masked sum_v h (-> hsum).
// ---------------------------------------------------------------------------
__global__ __launch_bounds__(256) void xattn_main(
    const float* __restrict__ qs1, const float* __restrict__ ks,
    const float* __restrict__ vs,  const unsigned char* __restrict__ sg,
    const float* __restrict__ b1,  const float* __restrict__ W2,
    const float* __restrict__ b2,  const __bf16* __restrict__ w1t,
    float* __restrict__ hsum_ws,   float* __restrict__ attn_ws)
{
  __shared__ __bf16 panel[2][16 * 8 * 512];    // 2 x 128 KB W1 panels
  __shared__ float  W2l[ND], b1l[ND], hsuml[ND];
  __shared__ float  sl[NV], red[NV];
  __shared__ unsigned int maskw[4];

  const int t    = threadIdx.x;
  const int lane = t & 31;
  const int w    = t >> 5;        // wave id = m-tile
  const int half = lane >> 4;
  const int bq   = blockIdx.x;    // b*128 + q
  const int b    = bq >> 7;

  for (int i = t; i < ND; i += 256) { W2l[i] = W2[i]; b1l[i] = b1[i]; hsuml[i] = 0.f; }
  if (t < 4) {
    unsigned int mw = 0;
    for (int j = 0; j < 32; ++j) mw |= (sg[b * NV + t * 32 + j] ? 1u : 0u) << j;
    maskw[t] = mw;
  }

  // Initial fill of panel[0] (nb block 0): 32 x 16B chunks per thread.
  {
    v4i* dst = (v4i*)panel[0];
#pragma unroll 4
    for (int i = 0; i < 32; ++i) {
      int c     = i * 256 + t;             // 16B chunk index, 8192 total
      int tileL = c >> 6, cw = c & 63;
      int kt    = tileL >> 3, ntl = tileL & 7;
      int gh    = ((kt * 32 + ntl) << 9) + (cw << 3);
      dst[c] = *(const v4i*)(w1t + gh);
    }
  }

  // Build register-resident A tiles: X[v][d] = q1[d] * ks[v][d] in bf16.
  // A layout (16-bit, 16x32): lane m = l&15; elem j -> k = half*16 + j
  const float* ksrow = ks  + ((size_t)(b * NV) + w * 16 + (lane & 15)) * ND;
  const float* qrow  = qs1 + (size_t)bq * ND;
  AV atile[16];
#pragma unroll
  for (int kt = 0; kt < 16; ++kt) {
    int d0 = kt * 32 + half * 16;
    F4 K0, K1, K2, K3, Q0, Q1, Q2, Q3;
    K0.v = *(const float4*)(ksrow + d0);      K1.v = *(const float4*)(ksrow + d0 + 4);
    K2.v = *(const float4*)(ksrow + d0 + 8); K3.v = *(const float4*)(ksrow + d0 + 12);
    Q0.v = *(const float4*)(qrow  + d0);      Q1.v = *(const float4*)(qrow  + d0 + 4);
    Q2.v = *(const float4*)(qrow  + d0 + 8); Q3.v = *(const float4*)(qrow  + d0 + 12);
#pragma unroll
    for (int i = 0; i < 4; ++i) {
      atile[kt].e[i]      = (__bf16)(K0.f[i] * Q0.f[i]);
      atile[kt].e[4 + i]  = (__bf16)(K1.f[i] * Q1.f[i]);
      atile[kt].e[8 + i]  = (__bf16)(K2.f[i] * Q2.f[i]);
      atile[kt].e[12 + i] = (__bf16)(K3.f[i] * Q3.f[i]);
    }
  }

  float sacc[8] = {0, 0, 0, 0, 0, 0, 0, 0};
  __syncthreads();   // maskw, W2l/b1l/hsuml, panel[0] all visible
  const unsigned int mword = maskw[(w * 16 + half * 8) >> 5];

  for (int nbk = 0; nbk < 4; ++nbk) {
    const int cur = nbk & 1;
    const __bf16* pnl  = panel[cur];
    v4i*          dstn = (v4i*)panel[cur ^ 1];
    const bool    stage = (nbk < 3);

    v8f acc[8];
#pragma unroll
    for (int nt = 0; nt < 8; ++nt) acc[nt] = v8f_zero();

#pragma unroll
    for (int kt = 0; kt < 16; ++kt) {
      // Interleave next-panel staging with the WMMA stream: 2 chunks per step.
      if (stage) {
#pragma unroll
        for (int u = 0; u < 2; ++u) {
          int c     = (kt * 2 + u) * 256 + t;
          int tileL = c >> 6, cw = c & 63;
          int kt2   = tileL >> 3, ntl = tileL & 7;
          int gh    = ((kt2 * 32 + (nbk + 1) * 8 + ntl) << 9) + (cw << 3);
          dstn[c] = *(const v4i*)(w1t + gh);
        }
      }
#pragma unroll
      for (int nt = 0; nt < 8; ++nt) {
        AV bv;
        const v4i* pb = (const v4i*)(pnl + (((kt << 3) + nt) << 9));
        bv.i[0] = pb[lane * 2];
        bv.i[1] = pb[lane * 2 + 1];
        acc[nt] = wmma_bf16(atile[kt], bv, acc[nt]);
      }
    }

    // Fused epilogue. C layout: reg r, lane l -> m = r + 8*half, n = l&15.
#pragma unroll
    for (int nt = 0; nt < 8; ++nt) {
      int   e   = nbk * 128 + nt * 16 + (lane & 15);
      float w2e = W2l[e], b1e = b1l[e];
      float hp  = 0.f;
#pragma unroll
      for (int r = 0; r < 8; ++r) {
        int   v = w * 16 + half * 8 + r;
        float h = fmaxf(acc[nt][r] + b1e, 0.f);
        sacc[r] = fmaf(h, w2e, sacc[r]);
        if (!((mword >> (v & 31)) & 1u)) hp += h;
      }
      hp += __shfl_xor(hp, 16);                 // combine v-halves (same e)
      if (lane < 16) atomicAdd(&hsuml[e], hp);  // ds_add_f32 across waves
    }
    __syncthreads();  // cur fully consumed; next panel fully written
  }

  // s[v]: reduce sacc across the 16 n-lanes in each half
#pragma unroll
  for (int r = 0; r < 8; ++r) {
    float s = sacc[r];
    s += __shfl_xor(s, 1); s += __shfl_xor(s, 2);
    s += __shfl_xor(s, 4); s += __shfl_xor(s, 8);
    sacc[r] = s;
  }
  if ((lane & 15) == 0) {
    int vb = w * 16 + half * 8;
#pragma unroll
    for (int r = 0; r < 8; ++r) sl[vb + r] = sacc[r];
  }
  __syncthreads();

  // softmax over v with mask
  float b2v = b2[0];
  if (t < NV) {
    float sv = fmaxf(sl[t] + b2v, 0.f);
    if ((maskw[t >> 5] >> (t & 31)) & 1u) sv = -1.0e9f;
    sl[t] = sv; red[t] = sv;
  }
  __syncthreads();
  for (int st = 64; st > 0; st >>= 1) {
    if (t < st) red[t] = fmaxf(red[t], red[t + st]);
    __syncthreads();
  }
  float mx = red[0];
  __syncthreads();
  if (t < NV) { float ev = __expf(sl[t] - mx); sl[t] = ev; red[t] = ev; }
  __syncthreads();
  for (int st = 64; st > 0; st >>= 1) {
    if (t < st) red[t] += red[t + st];
    __syncthreads();
  }
  float inv = 1.f / red[0];
  __syncthreads();
  if (t < NV) sl[t] *= inv;
  __syncthreads();

  // attention output  out[e] = sum_v w[v]*vs[b,v,e]  +  hsum writeback
  const float* vsb = vs + (size_t)b * NV * ND;
#pragma unroll
  for (int c = 0; c < 2; ++c) {
    int   e = t + c * 256;
    float a = 0.f;
    for (int v = 0; v < NV; ++v) a = fmaf(sl[v], vsb[(size_t)v * ND + e], a);
    attn_ws[(size_t)bq * ND + e] = a;
    hsum_ws[(size_t)bq * ND + e] = hsuml[e];
  }
}

// ---------------------------------------------------------------------------
// Kernel C: gate GEMM. mean(512x512, bf16) @ Wse(512x512, bf16 tiled) -> gate,
// out = sigmoid(relu(.+bse)) * attn. One WG per 16-row m-tile (32 WGs); wave w
// covers 64 output columns (4 n-tiles), full K.
// ---------------------------------------------------------------------------
__global__ __launch_bounds__(256) void xattn_gate(
    const float* __restrict__ hsum_ws, const float* __restrict__ attn_ws,
    const unsigned char* __restrict__ sg, const float* __restrict__ bse,
    const __bf16* __restrict__ wset, float* __restrict__ out)
{
  __shared__ __bf16 Abuf[16 * 520];   // 16 rows, padded pitch (bank spread)
  __shared__ float  dinv[NB];

  const int t    = threadIdx.x;
  const int lane = t & 31;
  const int w    = t >> 5;
  const int half = lane >> 4;
  const int mt   = blockIdx.x;        // m-tile: rows [mt*16, mt*16+16)

  if (t < NB) {
    int cnt = 0;
    for (int v = 0; v < NV; ++v) cnt += (sg[t * NV + v] == 0);
    dinv[t] = 1.f / (float)cnt;
  }
  __syncthreads();

  for (int i = t; i < 16 * ND; i += 256) {
    int rl = i >> 9, d = i & 511;
    int row = mt * 16 + rl;
    Abuf[rl * 520 + d] = (__bf16)(hsum_ws[(size_t)row * ND + d] * dinv[row >> 7]);
  }
  __syncthreads();

  v8f acc[4];
#pragma unroll
  for (int nt = 0; nt < 4; ++nt) acc[nt] = v8f_zero();

#pragma unroll
  for (int kt = 0; kt < 16; ++kt) {
    AV av;
    const __bf16* pa = Abuf + (lane & 15) * 520 + kt * 32 + half * 16;
    av.i[0] = *(const v4i*)pa;
    av.i[1] = *(const v4i*)(pa + 8);
#pragma unroll
    for (int nt = 0; nt < 4; ++nt) {
      AV bv;
      const __bf16* pb = wset + (((kt << 5) + (w * 4 + nt)) << 9) + lane * 16;
      bv.i[0] = *(const v4i*)pb;
      bv.i[1] = *(const v4i*)(pb + 8);
      acc[nt] = wmma_bf16(av, bv, acc[nt]);
    }
  }

#pragma unroll
  for (int nt = 0; nt < 4; ++nt) {
    int   e  = w * 64 + nt * 16 + (lane & 15);
    float be = bse[e];
#pragma unroll
    for (int r = 0; r < 8; ++r) {
      int   row = mt * 16 + half * 8 + r;
      float g   = fmaxf(acc[nt][r] + be, 0.f);
      g = 1.f / (1.f + __expf(-g));
      out[(size_t)row * ND + e] = g * attn_ws[(size_t)row * ND + e];
    }
  }
}

// ---------------------------------------------------------------------------
extern "C" void kernel_launch(void* const* d_in, const int* in_sizes, int n_in,
                              void* d_out, int out_size, void* d_ws, size_t ws_size,
                              hipStream_t stream)
{
  (void)in_sizes; (void)n_in; (void)out_size; (void)ws_size;

  const float*         qs1 = (const float*)d_in[0];
  // d_in[1] (qs_2) is unused by the reference
  const float*         ks  = (const float*)d_in[2];
  const float*         vs  = (const float*)d_in[3];
  const unsigned char* sg  = (const unsigned char*)d_in[4];
  const float*         W1  = (const float*)d_in[5];
  const float*         b1  = (const float*)d_in[6];
  const float*         Wse = (const float*)d_in[7];
  const float*         bse = (const float*)d_in[8];
  const float*         W2  = (const float*)d_in[9];
  const float*         b2  = (const float*)d_in[10];
  float*               out = (float*)d_out;

  // Workspace: [0,512K) W1 bf16 tiled | [512K,1M) Wse bf16 tiled |
  //            [1M,2M) hsum f32 [512][512] | [2M,3M) attn f32 [512][512]
  __bf16* w1t     = (__bf16*)d_ws;
  __bf16* wset    = (__bf16*)((char*)d_ws + (size_t)512 * 1024);
  float*  hsum_ws = (float*)((char*)d_ws + (size_t)1024 * 1024);
  float*  attn_ws = (float*)((char*)d_ws + (size_t)2048 * 1024);

  hipLaunchKernelGGL(prep_weights, dim3(2048), dim3(256), 0, stream,
                     W1, Wse, w1t, wset);
  hipLaunchKernelGGL(xattn_main, dim3(NB * NQ), dim3(256), 0, stream,
                     qs1, ks, vs, sg, b1, W2, b2, w1t, hsum_ws, attn_ws);
  hipLaunchKernelGGL(xattn_gate, dim3(32), dim3(256), 0, stream,
                     hsum_ws, attn_ws, sg, bse, wset, out);
}
